// GraphTransformer_82952998355942
// MI455X (gfx1250) — matrix operand order repaired and checked by reading, not compiled, not run on hardware
//
#include <hip/hip_runtime.h>
#include <hip/hip_bf16.h>

// ---------------------------------------------------------------------------
// GraphTransformer (PyG TransformerConv, concat=False) on MI455X / gfx1250.
// GEMMs use v_wmma_f32_16x16x32_bf16 (32 WMMAs/wave, B-fragment reuse x2);
// edge softmax uses fp32 global atomics with order-preserving uint max keys.
// ---------------------------------------------------------------------------

#define N_NODES 10000
#define E_EDGES 160000
#define HEADS   4
#define HID     128
#define KDIM    128     // input feature dim of every layer
#define QKV     512     // HEADS * HID
#define LAYERS  4

typedef __attribute__((ext_vector_type(16))) __bf16 v16bf;
typedef __attribute__((ext_vector_type(8)))  __bf16 v8bf;
typedef __attribute__((ext_vector_type(8)))  float  v8f;

// ---------------- helpers ----------------
__device__ __forceinline__ unsigned short f2bf(float f) {
    unsigned u = __float_as_uint(f);
    u += 0x7FFFu + ((u >> 16) & 1u);          // round to nearest even
    return (unsigned short)(u >> 16);
}
__device__ __forceinline__ unsigned fkey(float f) {   // order-preserving key
    unsigned u = __float_as_uint(f);
    return (u & 0x80000000u) ? ~u : (u | 0x80000000u);
}
__device__ __forceinline__ float funkey(unsigned k) {
    unsigned u = (k & 0x80000000u) ? (k & 0x7FFFFFFFu) : ~k;
    return __uint_as_float(u);
}

// ---------------- fp32 -> bf16 convert ----------------
__global__ void cvt_bf16(const float* __restrict__ in,
                         unsigned short* __restrict__ out, int n) {
    int tid = blockIdx.x * blockDim.x + threadIdx.x;
    if (tid < n) out[tid] = f2bf(in[tid]);
}

// ---------------- WMMA GEMM: C[MxNc] = A[Mx128](bf16) * W[128xNc](bf16) + bias
// 128 threads = 4 waves; block tile 128 rows x 64 cols; wave tile 32x64.
// Each B fragment (2x ds_load_b128) feeds TWO wmma ops (M-tile reuse).
__global__ __launch_bounds__(128)
void gemm_bf16_wmma(const unsigned short* __restrict__ A,
                    const unsigned short* __restrict__ W,
                    const float* __restrict__ bias,
                    float* __restrict__ C,
                    int Nrows, int Ncols) {
    __shared__ __align__(16) unsigned short Wlds[64][136]; // [col][k], padded

    const int t    = threadIdx.x;
    const int wave = t >> 5;
    const int lane = t & 31;
    const int colBase = blockIdx.x * 64;
    const int rowBase = blockIdx.y * 128 + wave * 32;

    // ---- stage weight tile transposed into LDS: Wlds[col][k] ----
    {
        const int chunk = t & 7;     // 8-column chunk
        const int k0    = t >> 3;    // 0..15
        #pragma unroll
        for (int it = 0; it < 8; ++it) {
            const int k = k0 + it * 16;                      // 0..127
            union { uint4 u; unsigned short s[8]; } d;
            d.u = *(const uint4*)(W + (size_t)k * Ncols + colBase + chunk * 8);
            #pragma unroll
            for (int j = 0; j < 8; ++j)
                Wlds[chunk * 8 + j][k] = d.s[j];
        }
    }
    __syncthreads();

    // ---- accumulators: 2 M-tiles x 4 N-tiles of 16x16 f32 ----
    v8f acc[2][4];
    #pragma unroll
    for (int rt = 0; rt < 2; ++rt)
        #pragma unroll
        for (int ct = 0; ct < 4; ++ct)
            #pragma unroll
            for (int r = 0; r < 8; ++r) acc[rt][ct][r] = 0.0f;

    // A-fragment addressing (16-bit A 16x32 layout):
    // lanes 0-15: row M=lane, K = {0..7, 16..23}; lanes 16-31: K = {8..15, 24..31}
    int rowA0 = rowBase + (lane & 15);
    int rowA1 = rowBase + 16 + (lane & 15);
    if (rowA0 >= Nrows) rowA0 = Nrows - 1;            // clamp: keep EXEC all-1s
    if (rowA1 >= Nrows) rowA1 = Nrows - 1;
    const unsigned short* arow0 = A + (size_t)rowA0 * KDIM;
    const unsigned short* arow1 = A + (size_t)rowA1 * KDIM;
    const int selA = (lane >> 4) << 3;                // 0 or 8
    const int selB = (lane >> 4) << 3;                // 0 or 8 (B: same K order as A)

    #pragma unroll
    for (int ks = 0; ks < 4; ++ks) {
        const int kb = ks * 32;
        v8bf a0lo = *(const v8bf*)(arow0 + kb + selA);
        v8bf a0hi = *(const v8bf*)(arow0 + kb + selA + 16);
        v8bf a1lo = *(const v8bf*)(arow1 + kb + selA);
        v8bf a1hi = *(const v8bf*)(arow1 + kb + selA + 16);
        v16bf af0 = __builtin_shufflevector(a0lo, a0hi,
            0,1,2,3,4,5,6,7,8,9,10,11,12,13,14,15);
        v16bf af1 = __builtin_shufflevector(a1lo, a1hi,
            0,1,2,3,4,5,6,7,8,9,10,11,12,13,14,15);

        #pragma unroll
        for (int ct = 0; ct < 4; ++ct) {
            const int colL = ct * 16 + (lane & 15);
            v8bf blo = *(const v8bf*)&Wlds[colL][kb + selB];
            v8bf bhi = *(const v8bf*)&Wlds[colL][kb + selB + 16];
            v16bf bfrag = __builtin_shufflevector(blo, bhi,
                0,1,2,3,4,5,6,7,8,9,10,11,12,13,14,15);
            acc[0][ct] = __builtin_amdgcn_wmma_f32_16x16x32_bf16(
                false, af0, false, bfrag, (short)0, acc[0][ct], false, false);
            acc[1][ct] = __builtin_amdgcn_wmma_f32_16x16x32_bf16(
                false, af1, false, bfrag, (short)0, acc[1][ct], false, false);
        }
    }

    // ---- store D + bias (C/D layout: VGPR r -> M=r (lanes 0-15) / M=r+8) ----
    #pragma unroll
    for (int rt = 0; rt < 2; ++rt) {
        #pragma unroll
        for (int ct = 0; ct < 4; ++ct) {
            const int col = colBase + ct * 16 + (lane & 15);
            const float b = bias[col];
            #pragma unroll
            for (int r = 0; r < 8; ++r) {
                const int row = rowBase + rt * 16 + r + ((lane >> 4) << 3);
                if (row < Nrows) C[(size_t)row * Ncols + col] = acc[rt][ct][r] + b;
            }
        }
    }
}

// ---------------- edge stage 1: logits + segment max ----------------
__global__ void edge_alpha(const float* __restrict__ q,
                           const float* __restrict__ k,
                           const int* __restrict__ ei,
                           float* __restrict__ alpha,
                           unsigned* __restrict__ mkey) {
    int tid = blockIdx.x * blockDim.x + threadIdx.x;
    if (tid >= E_EDGES * HEADS) return;
    const int e = tid >> 2, h = tid & 3;
    const int s = ei[e], d = ei[E_EDGES + e];
    const float4* qp = (const float4*)(q + (size_t)d * QKV + h * HID);
    const float4* kp = (const float4*)(k + (size_t)s * QKV + h * HID);
    float acc = 0.f;
    #pragma unroll
    for (int i = 0; i < HID / 4; ++i) {
        float4 a = qp[i], b = kp[i];
        acc += a.x * b.x + a.y * b.y + a.z * b.z + a.w * b.w;
    }
    acc *= 0.0883883476483184f;   // 1/sqrt(128)
    alpha[tid] = acc;
    atomicMax(&mkey[d * HEADS + h], fkey(acc));
}

// ---------------- edge stage 2: exp + segment sum ----------------
__global__ void edge_expsum(const int* __restrict__ ei,
                            const float* __restrict__ alpha,
                            const unsigned* __restrict__ mkey,
                            float* __restrict__ eexp,
                            float* __restrict__ z) {
    int tid = blockIdx.x * blockDim.x + threadIdx.x;
    if (tid >= E_EDGES * HEADS) return;
    const int e = tid >> 2, h = tid & 3;
    const int d = ei[E_EDGES + e];
    const float m = funkey(mkey[d * HEADS + h]);
    const float ex = __expf(alpha[tid] - m);
    eexp[tid] = ex;
    atomicAdd(&z[d * HEADS + h], ex);
}

// ---------------- edge stage 3: weighted message scatter ----------------
__global__ void edge_scatter(const int* __restrict__ ei,
                             const float* __restrict__ eexp,
                             const float* __restrict__ z,
                             const float* __restrict__ v,
                             float* __restrict__ accum) {
    int tid = blockIdx.x * blockDim.x + threadIdx.x;
    if (tid >= E_EDGES * HEADS * (HID / 4)) return;
    const int c4 = tid & 31;
    const int eh = tid >> 5;
    const int e = eh >> 2, h = eh & 3;
    const int s = ei[e], d = ei[E_EDGES + e];
    const float a = eexp[eh] / z[d * HEADS + h];
    const float4 vv = ((const float4*)(v + (size_t)s * QKV + h * HID))[c4];
    float* dst = accum + (size_t)d * QKV + h * HID + c4 * 4;
    atomicAdd(dst + 0, a * vv.x);
    atomicAdd(dst + 1, a * vv.y);
    atomicAdd(dst + 2, a * vv.z);
    atomicAdd(dst + 3, a * vv.w);
}

// ---------------- finalize: head mean + skip (+ReLU), emit bf16 or f32 ----
__global__ void finalize(const float* __restrict__ accum,
                         const float* __restrict__ skip,
                         float* __restrict__ outF,
                         unsigned short* __restrict__ outBF,
                         int doRelu) {
    int tid = blockIdx.x * blockDim.x + threadIdx.x;
    if (tid >= N_NODES * HID) return;
    const int n = tid >> 7, c = tid & 127;
    const float* ap = accum + (size_t)n * QKV + c;
    float o = 0.25f * (ap[0] + ap[HID] + ap[2 * HID] + ap[3 * HID]) + skip[tid];
    if (doRelu) o = fmaxf(o, 0.f);
    if (outF) outF[tid] = o;
    else      outBF[tid] = f2bf(o);
}

// ---------------------------------------------------------------------------
extern "C" void kernel_launch(void* const* d_in, const int* in_sizes, int n_in,
                              void* d_out, int out_size, void* d_ws, size_t ws_size,
                              hipStream_t stream) {
    (void)in_sizes; (void)n_in; (void)out_size; (void)ws_size;

    const float* x     = (const float*)d_in[0];
    const int*   ei    = (const int*)d_in[1];
    const float* Wq    = (const float*)d_in[2];
    const float* bq    = (const float*)d_in[3];
    const float* Wk    = (const float*)d_in[4];
    const float* bk    = (const float*)d_in[5];
    const float* Wv    = (const float*)d_in[6];
    const float* bv    = (const float*)d_in[7];
    const float* Wskip = (const float*)d_in[8];
    const float* bskip = (const float*)d_in[9];
    float* outF = (float*)d_out;

    // ---- workspace carve-up (256B aligned) ----
    char* base = (char*)d_ws;
    size_t off = 0;
    auto carve = [&](size_t bytes) -> char* {
        char* p = base + off;
        off += (bytes + 255) & ~(size_t)255;
        return p;
    };
    unsigned short* xbf  = (unsigned short*)carve((size_t)N_NODES * KDIM * 2);
    unsigned short* wqbf = (unsigned short*)carve((size_t)KDIM * QKV * 2);
    unsigned short* wkbf = (unsigned short*)carve((size_t)KDIM * QKV * 2);
    unsigned short* wvbf = (unsigned short*)carve((size_t)KDIM * QKV * 2);
    unsigned short* wsbf = (unsigned short*)carve((size_t)KDIM * HID * 2);
    float* q     = (float*)carve((size_t)N_NODES * QKV * 4);
    float* k     = (float*)carve((size_t)N_NODES * QKV * 4);
    float* v     = (float*)carve((size_t)N_NODES * QKV * 4);
    float* skip  = (float*)carve((size_t)N_NODES * HID * 4);
    float* alpha = (float*)carve((size_t)E_EDGES * HEADS * 4);
    float* eexp  = (float*)carve((size_t)E_EDGES * HEADS * 4);
    unsigned* mkey = (unsigned*)carve((size_t)N_NODES * HEADS * 4);
    float* z     = (float*)carve((size_t)N_NODES * HEADS * 4);
    float* accum = (float*)carve((size_t)N_NODES * QKV * 4);

    const int B = 256;
    const int nEH  = E_EDGES * HEADS;
    const int nSc  = E_EDGES * HEADS * (HID / 4);
    const int nNC  = N_NODES * HID;

    // input activations -> bf16
    cvt_bf16<<<(N_NODES * KDIM + B - 1) / B, B, 0, stream>>>(x, xbf, N_NODES * KDIM);

    dim3 gQ(QKV / 64, (N_NODES + 127) / 128);
    dim3 gS(HID / 64, (N_NODES + 127) / 128);

    for (int l = 0; l < LAYERS; ++l) {
        const int wqk = KDIM * QKV;        // 65536
        const int wsn = KDIM * HID;        // 16384
        cvt_bf16<<<(wqk + B - 1) / B, B, 0, stream>>>(Wq + (size_t)l * wqk, wqbf, wqk);
        cvt_bf16<<<(wqk + B - 1) / B, B, 0, stream>>>(Wk + (size_t)l * wqk, wkbf, wqk);
        cvt_bf16<<<(wqk + B - 1) / B, B, 0, stream>>>(Wv + (size_t)l * wqk, wvbf, wqk);
        cvt_bf16<<<(wsn + B - 1) / B, B, 0, stream>>>(Wskip + (size_t)l * wsn, wsbf, wsn);

        gemm_bf16_wmma<<<gQ, 128, 0, stream>>>(xbf, wqbf, bq + (size_t)l * QKV, q, N_NODES, QKV);
        gemm_bf16_wmma<<<gQ, 128, 0, stream>>>(xbf, wkbf, bk + (size_t)l * QKV, k, N_NODES, QKV);
        gemm_bf16_wmma<<<gQ, 128, 0, stream>>>(xbf, wvbf, bv + (size_t)l * QKV, v, N_NODES, QKV);
        gemm_bf16_wmma<<<gS, 128, 0, stream>>>(xbf, wsbf, bskip + (size_t)l * HID, skip, N_NODES, HID);

        hipMemsetAsync(mkey,  0, (size_t)N_NODES * HEADS * 4, stream); // key(0)=-NaN < all
        hipMemsetAsync(z,     0, (size_t)N_NODES * HEADS * 4, stream);
        hipMemsetAsync(accum, 0, (size_t)N_NODES * QKV * 4, stream);

        edge_alpha  <<<(nEH + B - 1) / B, B, 0, stream>>>(q, k, ei, alpha, mkey);
        edge_expsum <<<(nEH + B - 1) / B, B, 0, stream>>>(ei, alpha, mkey, eexp, z);
        edge_scatter<<<(nSc + B - 1) / B, B, 0, stream>>>(ei, eexp, z, v, accum);

        const int last = (l == LAYERS - 1);
        finalize<<<(nNC + B - 1) / B, B, 0, stream>>>(
            accum, skip, last ? outF : nullptr, xbf, last ? 0 : 1);
    }
}
